// MultiHeadAttention_35588099015388
// MI455X (gfx1250) — matrix operand order repaired and checked by reading, not compiled, not run on hardware
//
#include <hip/hip_runtime.h>

// ---------------------------------------------------------------------------
// MHA forward for B=2, S=2048, E=1024, H=16, D=64 on gfx1250 (MI455X).
// f16 WMMA (v_wmma_f32_16x16x32_f16) with f32 accumulation, flash-attention
// structure so the 1+ GB score matrix is never materialized in HBM.
// Round 4: 128x128 GEMM block tiles (8 WMMA / wave / K-step, 1.5 ds-loads per
// WMMA), branch-free clamped prefetch, async global->LDS staging retained.
// Workspace: Qh | Kh | Vh (f16, [B,H,S,D]) + attn-out (f16, [B,S,E]) = 32 MB.
// ---------------------------------------------------------------------------

typedef _Float16 h4   __attribute__((ext_vector_type(4)));
typedef _Float16 h8   __attribute__((ext_vector_type(8)));
typedef _Float16 v16h __attribute__((ext_vector_type(16)));
typedef float    v8f  __attribute__((ext_vector_type(8)));
typedef int      i4v  __attribute__((ext_vector_type(4)));

#if __has_builtin(__builtin_amdgcn_global_load_async_to_lds_b128)
#define HAVE_ASYNC_LDS 1
#else
#define HAVE_ASYNC_LDS 0
#endif

#if HAVE_ASYNC_LDS
typedef __attribute__((address_space(1))) i4v* gi4p;  // global int4*
typedef __attribute__((address_space(3))) i4v* li4p;  // LDS int4*

// One 16-byte async copy global -> LDS (tracked with ASYNCcnt).
__device__ __forceinline__ void async_copy16(const _Float16* gsrc,
                                             _Float16* ldst) {
  __builtin_amdgcn_global_load_async_to_lds_b128((gi4p)(void*)gsrc,
                                                 (li4p)(void*)ldst, 0, 0);
}
__device__ __forceinline__ void wait_async0() {
#if __has_builtin(__builtin_amdgcn_s_wait_asynccnt)
  __builtin_amdgcn_s_wait_asynccnt(0);
#else
  asm volatile("s_wait_asynccnt 0x0" ::: "memory");
#endif
}
#endif

// A-matrix 16x32 f16 fragment (ISA 7.12.2 table):
// lanes 0-15: row=lane, K=0..7 then 16..23 ; lanes 16-31: K=8..15 then 24..31
__device__ __forceinline__ v16h fragA(const _Float16* rowp, int lh) {
  const _Float16* p = rowp + lh * 8;
  h8 lo = *(const h8*)p;
  h8 hi = *(const h8*)(p + 16);
  return __builtin_shufflevector(lo, hi, 0,1,2,3,4,5,6,7,8,9,10,11,12,13,14,15);
}

// B-matrix 32x16 f16 fragment (same K order as fragA):
// lanes 0-15: col=lane, K=0..7 then 16..23 ; lanes 16-31: K=8..15 then 24..31
__device__ __forceinline__ v16h fragB(const _Float16* colp, int lh) {
  const _Float16* p = colp + lh * 8;
  h8 lo = *(const h8*)p;
  h8 hi = *(const h8*)(p + 16);
  return __builtin_shufflevector(lo, hi, 0,1,2,3,4,5,6,7,8,9,10,11,12,13,14,15);
}

__device__ __forceinline__ v8f wmma16(v16h a, v16h b, v8f c) {
  return __builtin_amdgcn_wmma_f32_16x16x32_f16(false, a, false, b, (short)0, c,
                                                false, false);
}

// ---------------------------------------------------------------------------
// Kernel 1: QKV projections.  Out = X[4096,1024] @ W[1024,1024]^T, written as
// f16 into [B,H,S,D] layout.  Q is pre-scaled by 1/sqrt(D) = 0.125 (exact).
// Block tile 128(M) x 128(N), K-step 32; 8 waves in 4(M)x2(N) grid, each wave
// computes a 32x64 sub-tile = 8 WMMAs per K-step.
// ---------------------------------------------------------------------------
__global__ __launch_bounds__(256) void proj_qkv_kernel(
    const float* __restrict__ qin, const float* __restrict__ kin,
    const float* __restrict__ vin, const float* __restrict__ wq,
    const float* __restrict__ wk, const float* __restrict__ wv,
    _Float16* __restrict__ qo, _Float16* __restrict__ ko,
    _Float16* __restrict__ vo) {
  const int z = blockIdx.z;
  const float* A = (z == 0) ? qin : (z == 1) ? kin : vin;
  const float* W = (z == 0) ? wq : (z == 1) ? wk : wv;
  _Float16* O   = (z == 0) ? qo : (z == 1) ? ko : vo;
  const float scale = (z == 0) ? 0.125f : 1.0f;

  __shared__ __align__(16) _Float16 sA[128 * 32];  // [row][k]
  __shared__ __align__(16) _Float16 sB[128 * 32];  // [n][k]  (W rows = B cols)

  const int t = threadIdx.x;
  const int lane = t & 31, wave = t >> 5;
  const int lr = lane & 15, lh = lane >> 4;
  const int mo = (wave >> 1) * 32, no = (wave & 1) * 64;
  const int blockM = blockIdx.y * 128;
  const int blockN = blockIdx.x * 128;

  const v8f vzero = {0.f, 0.f, 0.f, 0.f, 0.f, 0.f, 0.f, 0.f};
  v8f acc[2][4];
#pragma unroll
  for (int mt = 0; mt < 2; ++mt)
#pragma unroll
    for (int nt = 0; nt < 4; ++nt) acc[mt][nt] = vzero;

  for (int kb = 0; kb < 1024; kb += 32) {
    const int kpf = (kb + 32) & 1023;  // clamped (wrapping) prefetch target
    __syncthreads();
    // Stage A tile 128x32 f32 -> f16 (1024 float4 slots, 4 per thread)
#pragma unroll
    for (int i = 0; i < 4; ++i) {
      int slot = t + i * 256;
      int row = slot >> 3, kk = (slot & 7) * 4;
      const float* gp = A + (size_t)(blockM + row) * 1024 + kb + kk;
      __builtin_prefetch(A + (size_t)(blockM + row) * 1024 + kpf + kk, 0, 3);
      float4 f = *(const float4*)gp;
      h4 hv = {(_Float16)f.x, (_Float16)f.y, (_Float16)f.z, (_Float16)f.w};
      *(h4*)(sA + row * 32 + kk) = hv;
    }
    // Stage B tile 128x32 f32 -> f16 (1024 slots, 4 per thread)
#pragma unroll
    for (int i = 0; i < 4; ++i) {
      int slot = t + i * 256;
      int row = slot >> 3, kk = (slot & 7) * 4;
      const float* gp = W + (size_t)(blockN + row) * 1024 + kb + kk;
      __builtin_prefetch(W + (size_t)(blockN + row) * 1024 + kpf + kk, 0, 3);
      float4 f = *(const float4*)gp;
      h4 hv = {(_Float16)f.x, (_Float16)f.y, (_Float16)f.z, (_Float16)f.w};
      *(h4*)(sB + row * 32 + kk) = hv;
    }
    __syncthreads();

    v16h aF[2], bF[4];
#pragma unroll
    for (int mt = 0; mt < 2; ++mt)
      aF[mt] = fragA(sA + (mo + mt * 16 + lr) * 32, lh);
#pragma unroll
    for (int nt = 0; nt < 4; ++nt)
      bF[nt] = fragB(sB + (no + nt * 16 + lr) * 32, lh);
#pragma unroll
    for (int mt = 0; mt < 2; ++mt)
#pragma unroll
      for (int nt = 0; nt < 4; ++nt)
        acc[mt][nt] = wmma16(aF[mt], bF[nt], acc[mt][nt]);
  }

  // Epilogue: C layout (VGPR r -> row r (+8 for lanes 16-31), lane%16 -> col)
#pragma unroll
  for (int mt = 0; mt < 2; ++mt)
#pragma unroll
    for (int nt = 0; nt < 4; ++nt)
#pragma unroll
      for (int r = 0; r < 8; ++r) {
        int m = blockM + mo + mt * 16 + r + lh * 8;
        int n = blockN + no + nt * 16 + lr;
        int bb = m >> 11, s = m & 2047;   // m = b*S + s
        int hh = n >> 6, d = n & 63;      // n = h*D + d
        O[(((size_t)(bb * 16 + hh) * 2048 + s) * 64) + d] =
            (_Float16)(acc[mt][nt][r] * scale);
      }
}

// ---------------------------------------------------------------------------
// Kernel 2: flash attention per (b,h).  Block = 256 threads = 8 waves,
// each wave owns 32 query rows (block covers 256).  Key blocks of 32.
// K tile staged with GLOBAL_LOAD_ASYNC_TO_LDS_B128 (ASYNCcnt-tracked).
// Causal mask applied analytically; Q already carries the 1/sqrt(D) scale.
// ---------------------------------------------------------------------------
__global__ __launch_bounds__(256) void flash_attn_kernel(
    const _Float16* __restrict__ qh, const _Float16* __restrict__ kh,
    const _Float16* __restrict__ vh, _Float16* __restrict__ ao) {
  const int bh = blockIdx.y;
  const int b = bh >> 4, h = bh & 15;
  const int qBase = blockIdx.x * 256;
  const int t = threadIdx.x, lane = t & 31, wave = t >> 5;
  const int lr = lane & 15, lh = lane >> 4;
  const int qw = qBase + wave * 32;  // wave's first query row

  const _Float16* Q = qh + (size_t)bh * 2048 * 64;
  const _Float16* K = kh + (size_t)bh * 2048 * 64;
  const _Float16* V = vh + (size_t)bh * 2048 * 64;

  __shared__ __align__(16) _Float16 sK[32 * 64];      // [key][d]
  __shared__ __align__(16) _Float16 sVT[64 * 32];     // [d][key]
  __shared__ __align__(16) _Float16 sP[8][32 * 32];   // per-wave P tile

  // Preload Q fragments (rows qw..qw+31, D=64 split into 2 K-steps of 32)
  v16h qF[2][2];
#pragma unroll
  for (int mt = 0; mt < 2; ++mt) {
    int row = qw + mt * 16 + lr;
#pragma unroll
    for (int ks = 0; ks < 2; ++ks)
      qF[mt][ks] = fragA(Q + (size_t)row * 64 + ks * 32, lh);
  }

  const v8f vzero = {0.f, 0.f, 0.f, 0.f, 0.f, 0.f, 0.f, 0.f};
  v8f oAcc[2][4];
  float mS[2][8], lS[2][8];
#pragma unroll
  for (int mt = 0; mt < 2; ++mt) {
#pragma unroll
    for (int nt = 0; nt < 4; ++nt) oAcc[mt][nt] = vzero;
#pragma unroll
    for (int r = 0; r < 8; ++r) { mS[mt][r] = -1e30f; lS[mt][r] = 0.f; }
  }

  const int nKB = (qBase + 256) >> 5;  // causal: keys only up to block's rows
  for (int kbi = 0; kbi < nKB; ++kbi) {
    const int kb = kbi << 5;
    __syncthreads();  // previous iteration's LDS reads have retired
    {  // stage K tile 32x64 row-major: one 16B slot per thread
      int row = t >> 3, kk = (t & 7) * 8;
      const _Float16* gsrc = K + (size_t)(kb + row) * 64 + kk;
      _Float16* ldst = sK + row * 64 + kk;
#if HAVE_ASYNC_LDS
      async_copy16(gsrc, ldst);
#else
      *(h8*)ldst = *(const h8*)gsrc;
#endif
    }
    {  // stage V tile transposed: sVT[d][key] (register transpose)
      int key = t >> 3, d0 = (t & 7) * 8;
      h8 vv = *(const h8*)(V + (size_t)(kb + key) * 64 + d0);
#pragma unroll
      for (int j = 0; j < 8; ++j) sVT[(d0 + j) * 32 + key] = vv[j];
    }
#if HAVE_ASYNC_LDS
    wait_async0();  // this thread's async K bytes are in LDS
#endif
    __syncthreads();  // publish sK / sVT to all waves

    if (kb <= qw + 31) {  // wave-uniform predicate: EXEC stays all-ones
      // S = Q . K^T  (M=32, N=32 keys, K=64 over two steps)
      v8f sc[2][2];
#pragma unroll
      for (int mt = 0; mt < 2; ++mt)
#pragma unroll
        for (int nt = 0; nt < 2; ++nt) {
          v8f c = vzero;
#pragma unroll
          for (int ks = 0; ks < 2; ++ks) {
            v16h bF = fragB(sK + (nt * 16 + lr) * 64 + ks * 32, lh);
            c = wmma16(qF[mt][ks], bF, c);
          }
          sc[mt][nt] = c;
        }

      // Causal mask + online softmax (row reductions via 16-lane shuffles)
#pragma unroll
      for (int mt = 0; mt < 2; ++mt) {
#pragma unroll
        for (int nt = 0; nt < 2; ++nt) {
          int key = kb + nt * 16 + lr;
#pragma unroll
          for (int r = 0; r < 8; ++r) {
            int row = qw + mt * 16 + r + lh * 8;
            if (key > row) sc[mt][nt][r] = -1e30f;
          }
        }
#pragma unroll
        for (int r = 0; r < 8; ++r) {
          float mx = fmaxf(sc[mt][0][r], sc[mt][1][r]);
#pragma unroll
          for (int off = 8; off >= 1; off >>= 1)
            mx = fmaxf(mx, __shfl_xor(mx, off, 32));
          float mnew = fmaxf(mS[mt][r], mx);
          float corr = __expf(mS[mt][r] - mnew);
          float p0 = __expf(sc[mt][0][r] - mnew);
          float p1 = __expf(sc[mt][1][r] - mnew);
          sc[mt][0][r] = p0;
          sc[mt][1][r] = p1;
          float ps = p0 + p1;
#pragma unroll
          for (int off = 8; off >= 1; off >>= 1) ps += __shfl_xor(ps, off, 32);
          lS[mt][r] = lS[mt][r] * corr + ps;
          mS[mt][r] = mnew;
#pragma unroll
          for (int nt = 0; nt < 4; ++nt) oAcc[mt][nt][r] *= corr;
        }
      }

      // Reshape P: C-layout -> LDS (per-wave region; same-wave DS is in-order)
      _Float16* pw = sP[wave];
#pragma unroll
      for (int mt = 0; mt < 2; ++mt)
#pragma unroll
        for (int nt = 0; nt < 2; ++nt)
#pragma unroll
          for (int r = 0; r < 8; ++r)
            pw[(mt * 16 + r + lh * 8) * 32 + nt * 16 + lr] =
                (_Float16)sc[mt][nt][r];

      // O += P . V   (M=32, N=64 dims, K=32 keys, one WMMA step)
      v16h pF[2];
#pragma unroll
      for (int mt = 0; mt < 2; ++mt)
        pF[mt] = fragA(pw + (mt * 16 + lr) * 32, lh);
#pragma unroll
      for (int nt = 0; nt < 4; ++nt) {
        v16h vF = fragB(sVT + (nt * 16 + lr) * 32, lh);
#pragma unroll
        for (int mt = 0; mt < 2; ++mt)
          oAcc[mt][nt] = wmma16(pF[mt], vF, oAcc[mt][nt]);
      }
    }
  }

  // Normalize and write attn-out f16 [B,S,E] (row = b*S+s, col = h*64+d)
#pragma unroll
  for (int mt = 0; mt < 2; ++mt)
#pragma unroll
    for (int r = 0; r < 8; ++r) {
      float inv = 1.0f / lS[mt][r];
      int row = qw + mt * 16 + r + lh * 8;
      size_t base = ((size_t)b * 2048 + row) * 1024 + h * 64;
#pragma unroll
      for (int nt = 0; nt < 4; ++nt)
        ao[base + nt * 16 + lr] = (_Float16)(oAcc[mt][nt][r] * inv);
    }
}

// ---------------------------------------------------------------------------
// Kernel 3: output projection.  out[4096,1024] f32 = attn(f16) @ w_o^T.
// Block tile 128x128; A tile (f16) staged with async global->LDS copies.
// ---------------------------------------------------------------------------
__global__ __launch_bounds__(256) void out_proj_kernel(
    const _Float16* __restrict__ Ah, const float* __restrict__ W,
    float* __restrict__ out) {
  __shared__ __align__(16) _Float16 sA[128 * 32];
  __shared__ __align__(16) _Float16 sB[128 * 32];

  const int t = threadIdx.x;
  const int lane = t & 31, wave = t >> 5;
  const int lr = lane & 15, lh = lane >> 4;
  const int mo = (wave >> 1) * 32, no = (wave & 1) * 64;
  const int blockM = blockIdx.y * 128;
  const int blockN = blockIdx.x * 128;

  const v8f vzero = {0.f, 0.f, 0.f, 0.f, 0.f, 0.f, 0.f, 0.f};
  v8f acc[2][4];
#pragma unroll
  for (int mt = 0; mt < 2; ++mt)
#pragma unroll
    for (int nt = 0; nt < 4; ++nt) acc[mt][nt] = vzero;

  for (int kb = 0; kb < 1024; kb += 32) {
    const int kpf = (kb + 32) & 1023;
    __syncthreads();
    // A tile: 128x32 f16 = 512 x 16B slots, 2 per thread (async copies)
#pragma unroll
    for (int i = 0; i < 2; ++i) {
      int slot = t + i * 256;
      int row = slot >> 2, kk = (slot & 3) * 8;
      const _Float16* gsrc = Ah + (size_t)(blockM + row) * 1024 + kb + kk;
      _Float16* ldst = sA + row * 32 + kk;
#if HAVE_ASYNC_LDS
      async_copy16(gsrc, ldst);
#else
      *(h8*)ldst = *(const h8*)gsrc;
#endif
    }
    // B tile: 128x32 f32 -> f16 (1024 slots, 4 per thread)
#pragma unroll
    for (int i = 0; i < 4; ++i) {
      int slot = t + i * 256;
      int row = slot >> 3, kk = (slot & 7) * 4;
      const float* gp = W + (size_t)(blockN + row) * 1024 + kb + kk;
      __builtin_prefetch(W + (size_t)(blockN + row) * 1024 + kpf + kk, 0, 3);
      float4 f = *(const float4*)gp;
      h4 hv = {(_Float16)f.x, (_Float16)f.y, (_Float16)f.z, (_Float16)f.w};
      *(h4*)(sB + row * 32 + kk) = hv;
    }
#if HAVE_ASYNC_LDS
    wait_async0();
#endif
    __syncthreads();

    v16h aF[2], bF[4];
#pragma unroll
    for (int mt = 0; mt < 2; ++mt)
      aF[mt] = fragA(sA + (mo + mt * 16 + lr) * 32, lh);
#pragma unroll
    for (int nt = 0; nt < 4; ++nt)
      bF[nt] = fragB(sB + (no + nt * 16 + lr) * 32, lh);
#pragma unroll
    for (int mt = 0; mt < 2; ++mt)
#pragma unroll
      for (int nt = 0; nt < 4; ++nt)
        acc[mt][nt] = wmma16(aF[mt], bF[nt], acc[mt][nt]);
  }

#pragma unroll
  for (int mt = 0; mt < 2; ++mt)
#pragma unroll
    for (int nt = 0; nt < 4; ++nt)
#pragma unroll
      for (int r = 0; r < 8; ++r) {
        int m = blockM + mo + mt * 16 + r + lh * 8;
        int n = blockN + no + nt * 16 + lr;
        out[(size_t)m * 1024 + n] = acc[mt][nt][r];
      }
}

// ---------------------------------------------------------------------------
extern "C" void kernel_launch(void* const* d_in, const int* in_sizes, int n_in,
                              void* d_out, int out_size, void* d_ws,
                              size_t ws_size, hipStream_t stream) {
  (void)in_sizes; (void)n_in; (void)out_size; (void)ws_size;
  const float* q  = (const float*)d_in[0];
  const float* k  = (const float*)d_in[1];
  const float* v  = (const float*)d_in[2];
  // d_in[3] is the causal tril mask; handled analytically in flash_attn.
  const float* wq = (const float*)d_in[4];
  const float* wk = (const float*)d_in[5];
  const float* wv = (const float*)d_in[6];
  const float* wo = (const float*)d_in[7];
  float* out = (float*)d_out;

  const size_t NBH = (size_t)2 * 16 * 2048 * 64;  // 4,194,304 elems per tensor
  _Float16* qh = (_Float16*)d_ws;
  _Float16* kh = qh + NBH;
  _Float16* vh = kh + NBH;
  _Float16* ao = vh + NBH;  // total workspace use: 32 MB

  proj_qkv_kernel<<<dim3(8, 32, 3), 256, 0, stream>>>(q, k, v, wq, wk, wv,
                                                      qh, kh, vh);
  flash_attn_kernel<<<dim3(8, 32), 256, 0, stream>>>(qh, kh, vh, ao);
  out_proj_kernel<<<dim3(8, 32), 256, 0, stream>>>(ao, wo, out);
}
